// DiffusionModule_83837761618171
// MI455X (gfx1250) — compile-verified, matrix operand order checked by reading, not run on hardware
//
#include <hip/hip_runtime.h>
#include <hip/hip_bf16.h>

// ---------------------------------------------------------------------------
// MI455X (gfx1250). wave32, WMMA f32<-f16 16x16x32; f16 intermediates for the
// bandwidth-dominant tensors (biasAll / attention probs / qkv); async
// global->LDS copies (ASYNCcnt) for tiles that need no transpose/convert.
static constexpr int B_  = 2;
static constexpr int L_  = 768;
static constexpr int C_  = 256;
static constexpr int H_  = 8;
static constexpr int NB_ = 4;
static constexpr int M_  = B_ * L_;                    // 1536 token rows
static constexpr long long LL_ = (long long)L_ * L_;   // 589824
static constexpr int LDSP = 40;  // LDS row stride in halves (80B: 16B-aligned, 20-bank skew)

#define USE_ASYNC_COPY 1

typedef __attribute__((ext_vector_type(16))) _Float16 v16h;
typedef __attribute__((ext_vector_type(8)))  _Float16 v8h;
typedef __attribute__((ext_vector_type(4)))  _Float16 v4h;
typedef __attribute__((ext_vector_type(8)))  float    v8f;

#define WMMA16(a, b, c) \
  __builtin_amdgcn_wmma_f32_16x16x32_f16(false, (a), false, (b), (short)0, (c), false, false)

__device__ __forceinline__ float gelu_exact(float x) {
  return 0.5f * x * (1.0f + erff(x * 0.70710678118654752440f));
}

// 16B async copy global->LDS, tracked by ASYNCcnt (GLOBAL_LOAD_ASYNC_TO_LDS_B128).
// LDS flat pointers carry the wave-relative LDS byte offset in their low 32 bits.
__device__ __forceinline__ void async_copy_b128(void* lds_dst, const void* gsrc) {
#if USE_ASYNC_COPY
  unsigned lds = (unsigned)(size_t)lds_dst;
  asm volatile("global_load_async_to_lds_b128 %0, %1, off"
               :: "v"(lds), "v"(gsrc) : "memory");
#else
  *(v8h*)lds_dst = *(const v8h*)gsrc;
#endif
}
__device__ __forceinline__ void wait_async() {
#if USE_ASYNC_COPY
  asm volatile("s_wait_asynccnt 0x0" ::: "memory");
#endif
}

// Fragment load: one 32-halve LDS row -> v16h operand for 16x16x32 WMMA.
// Lane needs halves [lh*8, lh*8+8) and [16+lh*8, 16+lh*8+8): two ds_load_b128.
__device__ __forceinline__ v16h load_frag(const _Float16* row, int lh) {
  v8h lo = *(const v8h*)(row + lh * 8);
  v8h hi = *(const v8h*)(row + 16 + lh * 8);
  return __builtin_shufflevector(lo, hi, 0, 1, 2, 3, 4, 5, 6, 7,
                                 8, 9, 10, 11, 12, 13, 14, 15);
}

// float4 -> 4 packed halves
__device__ __forceinline__ v4h cvt4(const float4 f) {
  v4h r;
  r[0] = (_Float16)f.x; r[1] = (_Float16)f.y;
  r[2] = (_Float16)f.z; r[3] = (_Float16)f.w;
  return r;
}

// ---------------------------------------------------------------------------
// Generic WMMA GEMM: C[m,n] = epi(A[m,:K] * Bw[:K,n] + bias[n]) (+ resid)
// BM=128, BN=64, BK=32; 256 threads = 8 waves; wave owns a 16x64 strip.
// Double-buffered LDS ping-pong. EPI: 0=none, 1=GELU, 2=residual add.
// F16OUT: store _Float16 instead of float (used for the fused qkv buffer).
template <int EPI, bool HAS_BIAS, bool F16OUT>
__global__ __launch_bounds__(256) void k_gemm_t(
    const float* __restrict__ A, int lda,
    const float* __restrict__ Bw, int ldb,
    const float* __restrict__ bias,
    const float* __restrict__ resid,
    void* __restrict__ Cv, int ldc, int K)
{
  __shared__ _Float16 As[2][128][LDSP];   // A tile, row-major (m, k)
  __shared__ _Float16 Bt[2][64][LDSP];    // B tile TRANSPOSED: Bt[n][k]
  const int m0 = blockIdx.y * 128;
  const int n0 = blockIdx.x * 64;
  const int tid  = threadIdx.x;
  const int wave = tid >> 5;
  const int lane = tid & 31;
  const int lm = lane & 15;
  const int lh = lane >> 4;

  auto load_tiles = [&](int k0, int p) {
    #pragma unroll
    for (int i = 0; i < 4; ++i) {                 // A tile 128x32 = 1024 float4
      int idx = tid + i * 256;
      int r = idx >> 3, c4 = idx & 7;
      float4 f = *(const float4*)(A + (size_t)(m0 + r) * lda + k0 + c4 * 4);
      *(v4h*)(&As[p][r][c4 * 4]) = cvt4(f);
    }
    #pragma unroll
    for (int i = 0; i < 2; ++i) {                 // B tile 32x64 = 512 float4
      int idx = tid + i * 256;
      int r = idx >> 4, c4 = idx & 15;            // r = k row, c4*4 = n col
      float4 f = *(const float4*)(Bw + (size_t)(k0 + r) * ldb + n0 + c4 * 4);
      Bt[p][c4 * 4 + 0][r] = (_Float16)f.x;
      Bt[p][c4 * 4 + 1][r] = (_Float16)f.y;
      Bt[p][c4 * 4 + 2][r] = (_Float16)f.z;
      Bt[p][c4 * 4 + 3][r] = (_Float16)f.w;
    }
  };

  const int nk = K >> 5;
  load_tiles(0, 0);
  __syncthreads();

  v8f acc[4] = {};
  for (int kt = 0; kt < nk; ++kt) {
    const int p = kt & 1;
    if (kt + 1 < nk) load_tiles((kt + 1) << 5, p ^ 1);   // prefetch next tile
    v16h afrag = load_frag(&As[p][wave * 16 + lm][0], lh);
    #pragma unroll
    for (int nt = 0; nt < 4; ++nt) {
      v16h bfrag = load_frag(&Bt[p][nt * 16 + lm][0], lh);
      acc[nt] = WMMA16(afrag, bfrag, acc[nt]);
    }
    __syncthreads();
  }

  #pragma unroll
  for (int nt = 0; nt < 4; ++nt) {
    int n = n0 + nt * 16 + lm;
    const float bv = HAS_BIAS ? bias[n] : 0.f;
    #pragma unroll
    for (int r = 0; r < 8; ++r) {
      int m = m0 + wave * 16 + lh * 8 + r;
      float x = acc[nt][r] + bv;
      if (EPI == 1) x = gelu_exact(x);
      size_t o = (size_t)m * ldc + n;
      if (EPI == 2) x += resid[o];
      if (F16OUT) ((_Float16*)Cv)[o] = (_Float16)x;
      else        ((float*)Cv)[o] = x;
    }
  }
}

// ---------------------------------------------------------------------------
// Pair-bias for ALL 4 blocks in one pass (reads pair f32 exactly once, writes
// f16): biasAll[((nb*B+b)*H+h)*L*L + i*L + j] = sum_c pair[b,i,j,c]*pw[nb,c,h]
__global__ __launch_bounds__(256) void k_biasall(
    const float* __restrict__ pair, const float* __restrict__ pwAll,
    _Float16* __restrict__ biasAll)
{
  __shared__ _Float16 As[2][128][LDSP];
  __shared__ _Float16 Bt[2][32][LDSP];            // Bt[col][k], col = nb*8+h
  const size_t m0 = (size_t)blockIdx.x * 128;
  const int tid = threadIdx.x, wave = tid >> 5, lane = tid & 31;
  const int lm = lane & 15, lh = lane >> 4;

  auto load_tiles = [&](int k0, int p) {
    #pragma unroll
    for (int i = 0; i < 4; ++i) {                 // pair tile 128x32
      int idx = tid + i * 256;
      int r = idx >> 3, c4 = idx & 7;
      float4 f = *(const float4*)(pair + (m0 + r) * 64 + k0 + c4 * 4);
      *(v4h*)(&As[p][r][c4 * 4]) = cvt4(f);
    }
    #pragma unroll
    for (int i = 0; i < 4; ++i) {                 // pw tile: 32 cols x 32 k
      int idx = tid + i * 256;
      int col = idx & 31, k = idx >> 5;
      Bt[p][col][k] = (_Float16)pwAll[((size_t)(col >> 3) * 64 + k0 + k) * 8 + (col & 7)];
    }
  };

  load_tiles(0, 0);
  __syncthreads();
  v8f acc[2] = {};
  #pragma unroll
  for (int kt = 0; kt < 2; ++kt) {
    const int p = kt & 1;
    if (kt == 0) load_tiles(32, 1);
    v16h afrag = load_frag(&As[p][wave * 16 + lm][0], lh);
    #pragma unroll
    for (int nt = 0; nt < 2; ++nt) {
      v16h bfrag = load_frag(&Bt[p][nt * 16 + lm][0], lh);
      acc[nt] = WMMA16(afrag, bfrag, acc[nt]);
    }
    __syncthreads();
  }
  // lane's 8 outputs are contiguous in memory (M-dim == fastest dim here):
  // pack v8h and do one 16B store per tile.
  #pragma unroll
  for (int nt = 0; nt < 2; ++nt) {
    int col = nt * 16 + lm;
    int nb = col >> 3, h2 = col & 7;
    size_t row0 = m0 + wave * 16 + lh * 8;        // = b*L*L + i*L + j
    size_t b = row0 / (size_t)LL_;
    size_t rem0 = row0 - b * (size_t)LL_;
    v8h pk;
    #pragma unroll
    for (int r = 0; r < 8; ++r) pk[r] = (_Float16)acc[nt][r];
    *(v8h*)(biasAll + (((size_t)nb * B_ + b) * H_ + h2) * (size_t)LL_ + rem0) = pk;
  }
}

// ---------------------------------------------------------------------------
// Attention scores: S[b,h,i,j] = scale * (q_i . k_j) + biasBlk[b,h,i,j]
// grid: x = j-tile (L/64), y = i-tile (L/128), z = b*H + h
// qkv is f16 -> Q and K tiles are byte-identical to LDS layout: async copy.
__global__ __launch_bounds__(256) void k_attn_score(
    const _Float16* __restrict__ qkv, const _Float16* __restrict__ biasBlk,
    float* __restrict__ S, float scale)
{
  __shared__ _Float16 Qs[128][LDSP];
  __shared__ _Float16 Ks[64][LDSP];               // Ks[j][d]
  const int bh = blockIdx.z, b = bh >> 3, hh = bh & 7;
  const int i0 = blockIdx.y * 128, j0 = blockIdx.x * 64;
  const int tid = threadIdx.x, wave = tid >> 5, lane = tid & 31;
  const int lm = lane & 15, lh = lane >> 4;

  #pragma unroll
  for (int i = 0; i < 2; ++i) {                   // Q tile 128x32h = 512 x 16B
    int idx = tid + i * 256;
    int r = idx >> 2, c2 = idx & 3;
    async_copy_b128(&Qs[r][c2 * 8],
                    qkv + (size_t)(b * L_ + i0 + r) * 768 + hh * 32 + c2 * 8);
  }
  {                                               // K tile 64x32h = 256 x 16B
    int r = tid >> 2, c2 = tid & 3;
    async_copy_b128(&Ks[r][c2 * 8],
                    qkv + (size_t)(b * L_ + j0 + r) * 768 + 256 + hh * 32 + c2 * 8);
  }
  wait_async();
  __syncthreads();

  v16h afrag = load_frag(&Qs[wave * 16 + lm][0], lh);
  v8f acc[4] = {};
  #pragma unroll
  for (int nt = 0; nt < 4; ++nt) {
    v16h bfrag = load_frag(&Ks[nt * 16 + lm][0], lh);
    acc[nt] = WMMA16(afrag, bfrag, acc[nt]);
  }
  #pragma unroll
  for (int nt = 0; nt < 4; ++nt) {
    int j = j0 + nt * 16 + lm;
    #pragma unroll
    for (int r = 0; r < 8; ++r) {
      int i = i0 + wave * 16 + lh * 8 + r;
      size_t o = ((size_t)bh * L_ + i) * L_ + j;
      S[o] = scale * acc[nt][r] + (float)biasBlk[o];
    }
  }
}

// ---------------------------------------------------------------------------
// Row softmax: read f32 scores, write f16 probabilities (WMMA-ready).
// One wave32 per 768-wide row; 8 rows per block.
__global__ __launch_bounds__(256) void k_softmax(
    const float* __restrict__ S, _Float16* __restrict__ P)
{
  const int row  = blockIdx.x * 8 + (threadIdx.x >> 5);
  const int lane = threadIdx.x & 31;
  const float* p = S + (size_t)row * L_;
  _Float16* q = P + (size_t)row * L_;
  float v[24];
  float mx = -3.4e38f;
  #pragma unroll
  for (int i = 0; i < 24; ++i) { v[i] = p[lane + i * 32]; mx = fmaxf(mx, v[i]); }
  #pragma unroll
  for (int off = 16; off; off >>= 1) mx = fmaxf(mx, __shfl_xor(mx, off, 32));
  float s = 0.f;
  #pragma unroll
  for (int i = 0; i < 24; ++i) { v[i] = expf(v[i] - mx); s += v[i]; }
  #pragma unroll
  for (int off = 16; off; off >>= 1) s += __shfl_xor(s, off, 32);
  float inv = 1.0f / s;
  #pragma unroll
  for (int i = 0; i < 24; ++i) q[lane + i * 32] = (_Float16)(v[i] * inv);
}

// ---------------------------------------------------------------------------
// O[b,i, h*32+n] = sum_j P[b,h,i,j] * v[b,j,h,n]   (K = L = 768, N = 32)
// Double-buffered; P tiles (f16, no transpose) arrive via async copy.
__global__ __launch_bounds__(256) void k_attn_av(
    const _Float16* __restrict__ P, const _Float16* __restrict__ qkv,
    float* __restrict__ attnout)
{
  __shared__ _Float16 Ps[2][128][LDSP];
  __shared__ _Float16 Vt[2][32][LDSP];            // Vt[n][k] (transposed)
  const int bh = blockIdx.z, b = bh >> 3, hh = bh & 7;
  const int i0 = blockIdx.y * 128;
  const int tid = threadIdx.x, wave = tid >> 5, lane = tid & 31;
  const int lm = lane & 15, lh = lane >> 4;

  auto load_tiles = [&](int k0, int p) {
    #pragma unroll
    for (int i = 0; i < 2; ++i) {                 // P tile 128x32h = 512 x 16B
      int idx = tid + i * 256;
      int r = idx >> 2, c2 = idx & 3;
      async_copy_b128(&Ps[p][r][c2 * 8],
                      P + ((size_t)bh * L_ + i0 + r) * L_ + k0 + c2 * 8);
    }
    if (tid < 128) {                              // V tile 32x32h, transposed
      int r = tid >> 2, c2 = tid & 3;             // r = k row, c2*8 = n col
      v8h f = *(const v8h*)(qkv + (size_t)(b * L_ + k0 + r) * 768 + 512 + hh * 32 + c2 * 8);
      #pragma unroll
      for (int j = 0; j < 8; ++j) Vt[p][c2 * 8 + j][r] = f[j];
    }
  };

  load_tiles(0, 0);
  wait_async();
  __syncthreads();
  v8f acc[2] = {};
  const int nk = L_ / 32;                          // 24
  for (int kt = 0; kt < nk; ++kt) {
    const int p = kt & 1;
    if (kt + 1 < nk) load_tiles((kt + 1) << 5, p ^ 1);
    v16h afrag = load_frag(&Ps[p][wave * 16 + lm][0], lh);
    #pragma unroll
    for (int nt = 0; nt < 2; ++nt) {
      v16h bfrag = load_frag(&Vt[p][nt * 16 + lm][0], lh);
      acc[nt] = WMMA16(afrag, bfrag, acc[nt]);
    }
    wait_async();
    __syncthreads();
  }
  #pragma unroll
  for (int nt = 0; nt < 2; ++nt) {
    int n = nt * 16 + lm;
    #pragma unroll
    for (int r = 0; r < 8; ++r) {
      int i = i0 + wave * 16 + lh * 8 + r;
      attnout[(size_t)(b * L_ + i) * C_ + hh * 32 + n] = acc[nt][r];
    }
  }
}

// ---------------------------------------------------------------------------
// adaLN: out = (LN(h)*g+beta)*(1+scale)+shift. One block per row.
__global__ __launch_bounds__(256) void k_adaln(
    const float* __restrict__ h, const float* __restrict__ ss,
    const float* __restrict__ g, const float* __restrict__ beta,
    float* __restrict__ out)
{
  __shared__ float red[8];
  const int m = blockIdx.x, c = threadIdx.x;
  const int b = m / L_;
  const float x = h[(size_t)m * C_ + c];
  float v = x;
  #pragma unroll
  for (int off = 16; off; off >>= 1) v += __shfl_xor(v, off, 32);
  if ((c & 31) == 0) red[c >> 5] = v;
  __syncthreads();
  float mu = 0.f;
  #pragma unroll
  for (int i = 0; i < 8; ++i) mu += red[i];
  mu *= (1.0f / C_);
  __syncthreads();
  const float d = x - mu;
  v = d * d;
  #pragma unroll
  for (int off = 16; off; off >>= 1) v += __shfl_xor(v, off, 32);
  if ((c & 31) == 0) red[c >> 5] = v;
  __syncthreads();
  float var = 0.f;
  #pragma unroll
  for (int i = 0; i < 8; ++i) var += red[i];
  var *= (1.0f / C_);
  const float y  = d * rsqrtf(var + 1e-5f) * g[c] + beta[c];
  const float sc = ss[b * 512 + c];
  const float sh = ss[b * 512 + 256 + c];
  out[(size_t)m * C_ + c] = y * (1.0f + sc) + sh;
}

// ---------------------------------------------------------------------------
// Small/tiny matmuls (time MLP, adaLN cond proj, output head): thread/(m,n).
__global__ void k_small_mm(const float* __restrict__ A, int lda,
                           const float* __restrict__ Bw, int ldb,
                           const float* __restrict__ bias,
                           float* __restrict__ out, int ldo,
                           int M, int N, int K, int dogelu)
{
  int idx = blockIdx.x * blockDim.x + threadIdx.x;
  if (idx >= M * N) return;
  int m = idx / N, n = idx % N;
  float s = bias ? bias[n] : 0.f;
  for (int k = 0; k < K; ++k) s += A[(size_t)m * lda + k] * Bw[(size_t)k * ldb + n];
  if (dogelu) s = gelu_exact(s);
  out[(size_t)m * ldo + n] = s;
}

// ---------------------------------------------------------------------------
__global__ void k_timeemb(const float* __restrict__ t, float* __restrict__ temb)
{
  int idx = blockIdx.x * blockDim.x + threadIdx.x;
  if (idx >= B_ * C_) return;
  int b = idx >> 8, j = idx & 255;
  int jj = j & 127;
  float freq = expf(-9.210340371976184f * (float)jj / 128.0f);
  float a = t[b] * freq;
  temb[idx] = (j < 128) ? cosf(a) : sinf(a);
}

// frame embedding (K=12) + both biases; the K=256 part is added by k_gemm epi=2
__global__ void k_frame_embed(const float* __restrict__ rots,
                              const float* __restrict__ trans,
                              const float* __restrict__ frame_w,
                              const float* __restrict__ frame_b,
                              const float* __restrict__ single_b,
                              float* __restrict__ h)
{
  int idx = blockIdx.x * blockDim.x + threadIdx.x;
  if (idx >= M_ * C_) return;
  int m = idx >> 8, n = idx & 255;
  float s = frame_b[n] + single_b[n];
  #pragma unroll
  for (int k = 0; k < 9; ++k) s += rots[m * 9 + k] * frame_w[k * C_ + n];
  #pragma unroll
  for (int k = 0; k < 3; ++k) s += trans[m * 3 + k] * frame_w[(9 + k) * C_ + n];
  h[idx] = s;
}

// axis-angle -> R, compose frames, write (B,L,12)
__global__ void k_frame_update(const float* __restrict__ rots,
                               const float* __restrict__ trans,
                               const float* __restrict__ corr,
                               float* __restrict__ out)
{
  int m = blockIdx.x * blockDim.x + threadIdx.x;
  if (m >= M_) return;
  const float* c = corr + (size_t)m * 6;
  float vx = c[0], vy = c[1], vz = c[2];
  float n = sqrtf(vx * vx + vy * vy + vz * vz);
  float inv = 1.0f / (n + 1e-8f);
  float ax = vx * inv, ay = vy * inv, az = vz * inv;
  float s = sinf(n), cc = 1.0f - cosf(n);
  float R[9];
  R[0] = 1.f - cc * (ay * ay + az * az);
  R[1] = -s * az + cc * ax * ay;
  R[2] =  s * ay + cc * ax * az;
  R[3] =  s * az + cc * ax * ay;
  R[4] = 1.f - cc * (ax * ax + az * az);
  R[5] = -s * ax + cc * ay * az;
  R[6] = -s * ay + cc * ax * az;
  R[7] =  s * ax + cc * ay * az;
  R[8] = 1.f - cc * (ax * ax + ay * ay);
  const float* Rin = rots + (size_t)m * 9;
  float* o = out + (size_t)m * 12;
  #pragma unroll
  for (int i = 0; i < 3; ++i)
    #pragma unroll
    for (int j = 0; j < 3; ++j)
      o[i * 3 + j] = Rin[i * 3 + 0] * R[0 * 3 + j] +
                     Rin[i * 3 + 1] * R[1 * 3 + j] +
                     Rin[i * 3 + 2] * R[2 * 3 + j];
  float tx = c[3], ty = c[4], tz = c[5];
  #pragma unroll
  for (int i = 0; i < 3; ++i)
    o[9 + i] = Rin[i * 3 + 0] * tx + Rin[i * 3 + 1] * ty + Rin[i * 3 + 2] * tz
             + trans[(size_t)m * 3 + i];
}

// ---------------------------------------------------------------------------
extern "C" void kernel_launch(void* const* d_in, const int* in_sizes, int n_in,
                              void* d_out, int out_size, void* d_ws, size_t ws_size,
                              hipStream_t stream) {
  const float* rots     = (const float*)d_in[0];
  const float* trans    = (const float*)d_in[1];
  const float* t        = (const float*)d_in[2];
  const float* single   = (const float*)d_in[3];
  const float* pair     = (const float*)d_in[4];
  const float* frame_w  = (const float*)d_in[5];
  const float* frame_b  = (const float*)d_in[6];
  const float* single_w = (const float*)d_in[7];
  const float* single_b = (const float*)d_in[8];
  const float* tw1      = (const float*)d_in[9];
  const float* tb1      = (const float*)d_in[10];
  const float* tw2      = (const float*)d_in[11];
  const float* tb2      = (const float*)d_in[12];
  const float* out_w    = (const float*)d_in[13];
  const float* out_b    = (const float*)d_in[14];
  const float* ag1      = (const float*)d_in[15];
  const float* abeta1   = (const float*)d_in[16];
  const float* apw1     = (const float*)d_in[17];
  const float* apb1     = (const float*)d_in[18];
  const float* wq       = (const float*)d_in[19];
  const float* wk       = (const float*)d_in[20];
  const float* wv       = (const float*)d_in[21];
  const float* pw       = (const float*)d_in[22];
  const float* wo       = (const float*)d_in[23];
  const float* wob      = (const float*)d_in[24];
  const float* ag2      = (const float*)d_in[25];
  const float* abeta2   = (const float*)d_in[26];
  const float* apw2     = (const float*)d_in[27];
  const float* apb2     = (const float*)d_in[28];
  const float* fw1      = (const float*)d_in[29];
  const float* fb1      = (const float*)d_in[30];
  const float* fw2      = (const float*)d_in[31];
  const float* fb2      = (const float*)d_in[32];
  float* outp = (float*)d_out;

  // workspace carve-up (float units); ~130 MB total after f16 intermediates
  float* ws = (float*)d_ws;
  size_t off = 0;
  auto alloc = [&](size_t n) { float* p = ws + off; off += n; return p; };
  float* temb      = alloc((size_t)B_ * C_);
  float* tc1       = alloc((size_t)B_ * 4 * C_);
  float* time_cond = alloc((size_t)B_ * C_);
  float* ss        = alloc((size_t)B_ * 2 * C_);
  float* h         = alloc((size_t)M_ * C_);
  float* hh        = alloc((size_t)M_ * C_);
  float* attnout   = alloc((size_t)M_ * C_);
  float* ffmid     = alloc((size_t)M_ * 4 * C_);
  float* corr      = alloc((size_t)M_ * 6);
  float* S         = alloc((size_t)B_ * H_ * (size_t)LL_);              // 37.7 MB f32
  _Float16* qkvh   = (_Float16*)alloc((size_t)M_ * 3 * C_ / 2);         // f16
  _Float16* Pf16   = (_Float16*)alloc((size_t)B_ * H_ * (size_t)LL_ / 2);      // 18.9 MB
  _Float16* biasAll= (_Float16*)alloc((size_t)NB_ * B_ * H_ * (size_t)LL_ / 2);// 75.5 MB

  const float scale = 0.17677669529663687f;  // 1/sqrt(HD)

  // time embedding MLP (tiny)
  k_timeemb<<<2, 256, 0, stream>>>(t, temb);
  k_small_mm<<<(B_ * 4 * C_ + 255) / 256, 256, 0, stream>>>(
      temb, C_, tw1, 4 * C_, tb1, tc1, 4 * C_, B_, 4 * C_, C_, 1);
  k_small_mm<<<(B_ * C_ + 255) / 256, 256, 0, stream>>>(
      tc1, 4 * C_, tw2, C_, tb2, time_cond, C_, B_, C_, 4 * C_, 0);

  // token embedding: frame part (K=12) + single @ single_w via WMMA residual-add
  k_frame_embed<<<(M_ * C_ + 255) / 256, 256, 0, stream>>>(
      rots, trans, frame_w, frame_b, single_b, h);
  k_gemm_t<2, false, false><<<dim3(C_ / 64, M_ / 128), 256, 0, stream>>>(
      single, C_, single_w, C_, nullptr, h, h, C_, C_);

  // pair bias for all 4 blocks: read pair ONCE (302 MB), write 75 MB f16
  k_biasall<<<(int)((B_ * LL_) / 128), 256, 0, stream>>>(pair, pw, biasAll);

  for (int i = 0; i < NB_; ++i) {
    // adaLN 1
    k_small_mm<<<(B_ * 2 * C_ + 255) / 256, 256, 0, stream>>>(
        time_cond, C_, apw1 + (size_t)i * C_ * 2 * C_, 2 * C_,
        apb1 + (size_t)i * 2 * C_, ss, 2 * C_, B_, 2 * C_, C_, 0);
    k_adaln<<<M_, 256, 0, stream>>>(h, ss, ag1 + (size_t)i * C_,
                                    abeta1 + (size_t)i * C_, hh);
    // fused QKV into one (M,768) f16 buffer (WMMA-ready for the attn kernels)
    k_gemm_t<0, false, true><<<dim3(C_ / 64, M_ / 128), 256, 0, stream>>>(
        hh, C_, wq + (size_t)i * C_ * C_, C_, nullptr, nullptr, qkvh + 0,   768, C_);
    k_gemm_t<0, false, true><<<dim3(C_ / 64, M_ / 128), 256, 0, stream>>>(
        hh, C_, wk + (size_t)i * C_ * C_, C_, nullptr, nullptr, qkvh + 256, 768, C_);
    k_gemm_t<0, false, true><<<dim3(C_ / 64, M_ / 128), 256, 0, stream>>>(
        hh, C_, wv + (size_t)i * C_ * C_, C_, nullptr, nullptr, qkvh + 512, 768, C_);
    // attention: scores (+precomputed f16 bias), softmax -> f16 probs, P@V
    const _Float16* biasBlk = biasAll + (size_t)i * B_ * H_ * LL_;
    k_attn_score<<<dim3(L_ / 64, L_ / 128, B_ * H_), 256, 0, stream>>>(
        qkvh, biasBlk, S, scale);
    k_softmax<<<(B_ * H_ * L_) / 8, 256, 0, stream>>>(S, Pf16);
    k_attn_av<<<dim3(1, L_ / 128, B_ * H_), 256, 0, stream>>>(Pf16, qkvh, attnout);
    // h += attnout @ wo + wob
    k_gemm_t<2, true, false><<<dim3(C_ / 64, M_ / 128), 256, 0, stream>>>(
        attnout, C_, wo + (size_t)i * C_ * C_, C_, wob + (size_t)i * C_,
        h, h, C_, C_);
    // adaLN 2 + FFN
    k_small_mm<<<(B_ * 2 * C_ + 255) / 256, 256, 0, stream>>>(
        time_cond, C_, apw2 + (size_t)i * C_ * 2 * C_, 2 * C_,
        apb2 + (size_t)i * 2 * C_, ss, 2 * C_, B_, 2 * C_, C_, 0);
    k_adaln<<<M_, 256, 0, stream>>>(h, ss, ag2 + (size_t)i * C_,
                                    abeta2 + (size_t)i * C_, hh);
    k_gemm_t<1, true, false><<<dim3((4 * C_) / 64, M_ / 128), 256, 0, stream>>>(
        hh, C_, fw1 + (size_t)i * C_ * 4 * C_, 4 * C_, fb1 + (size_t)i * 4 * C_,
        nullptr, ffmid, 4 * C_, C_);
    k_gemm_t<2, true, false><<<dim3(C_ / 64, M_ / 128), 256, 0, stream>>>(
        ffmid, 4 * C_, fw2 + (size_t)i * 4 * C_ * C_, C_, fb2 + (size_t)i * C_,
        h, h, C_, 4 * C_);
  }

  // output head + frame composition
  k_small_mm<<<(M_ * 6 + 255) / 256, 256, 0, stream>>>(
      h, C_, out_w, 6, out_b, corr, 6, M_, 6, C_, 0);
  k_frame_update<<<(M_ + 255) / 256, 256, 0, stream>>>(rots, trans, corr, outp);
}
